// SlidingWindowAttention_77000173682697
// MI455X (gfx1250) — matrix operand order repaired and checked by reading, not compiled, not run on hardware
//
#include <hip/hip_runtime.h>
#include <hip/hip_bf16.h>

// Problem constants (match reference)
#define B_    2
#define T_    2048
#define HID_  2048
#define NH_   16
#define NKV_  4
#define D_    128
#define WIN_  512
#define MROWS (B_ * T_)                 // 4096
#define SCALE_ 0.08838834764831845f    // 128^-0.5

typedef __bf16 bf16_t;
typedef bf16_t v16bf __attribute__((ext_vector_type(16)));
typedef float  v8f   __attribute__((ext_vector_type(8)));
typedef unsigned int u32x4 __attribute__((ext_vector_type(4)));
typedef unsigned int u32x8 __attribute__((ext_vector_type(8)));

union FragBF { v16bf v; uint4 u[2]; };

__device__ __forceinline__ unsigned short f2bf(float f) {
  unsigned int u = __float_as_uint(f);
  return (unsigned short)((u + 0x7FFFu + ((u >> 16) & 1u)) >> 16);  // RNE
}

__device__ __forceinline__ v8f wmma_bf16(v16bf a, v16bf b, v8f c) {
  // D(16x16,f32) = A(16x32,bf16) * B(32x16,bf16) + C
  return __builtin_amdgcn_wmma_f32_16x16x32_bf16(
      false, a, false, b, (short)0, c, false, false);
}

// LDS byte offset of a __shared__ object (generic -> AS3 -> int)
__device__ __forceinline__ unsigned lds_offset_of(const void* p) {
  return (unsigned)(unsigned long long)
      (__attribute__((address_space(3))) const void*)p;
}

// ---------------------------------------------------------------------------
// fp32 -> bf16 cast (grid-stride)
// ---------------------------------------------------------------------------
__global__ void cast_bf16_kernel(const float* __restrict__ s,
                                 unsigned short* __restrict__ d, int n) {
  for (int i = blockIdx.x * blockDim.x + threadIdx.x; i < n;
       i += gridDim.x * blockDim.x)
    d[i] = f2bf(s[i]);
}

// ---------------------------------------------------------------------------
// WMMA bf16 GEMM: C[M,N] (f32) = A[M,K] * B[K,N]  (A,B bf16 row-major)
// Block: 256 thr (8 waves), tile 128x64, wave tile 32x32, K-step 32.
// Software-pipelined with LDS double buffering (one barrier / K-step).
// ---------------------------------------------------------------------------
__global__ __launch_bounds__(256) void gemm_bf16_wmma(
    const unsigned short* __restrict__ A, const unsigned short* __restrict__ Bm,
    float* __restrict__ C, int M, int N, int K) {
  __shared__ unsigned short sA[2][128 * 32];   // [buf][row][k]
  __shared__ unsigned short sBt[2][64 * 32];   // [buf][n][k] (transposed)

  const int tid  = threadIdx.x;
  const int lane = tid & 31, wave = tid >> 5;
  const int half = lane >> 4, ln = lane & 15;
  const int wm = wave >> 1, wn = wave & 1;
  const int rowBase = blockIdx.y * 128;
  const int colBase = blockIdx.x * 64;

  v8f zf = {};
  v8f acc[2][2];
#pragma unroll
  for (int i = 0; i < 2; i++)
#pragma unroll
    for (int j = 0; j < 2; j++) acc[i][j] = zf;

  const int ar = tid >> 1, ac = (tid & 1) * 16;   // A stager: 16 bf16 each
  const int bk = tid >> 3, bn = (tid & 7) * 8;    // B stager: 8 bf16 each

  // Prologue: stage tile 0 into buffer 0
  {
    const uint4* ag = (const uint4*)(A + (size_t)(rowBase + ar) * K + ac);
    *(uint4*)&sA[0][ar * 32 + ac]     = ag[0];
    *(uint4*)&sA[0][ar * 32 + ac + 8] = ag[1];
    uint4 bv = *(const uint4*)(Bm + (size_t)bk * N + colBase + bn);
    const unsigned short* bs = (const unsigned short*)&bv;
#pragma unroll
    for (int j = 0; j < 8; j++) sBt[0][(bn + j) * 32 + bk] = bs[j];
  }
  int buf = 0;

  for (int k0 = 0; k0 < K; k0 += 32) {
    __syncthreads();   // staged tile `buf` visible; prior readers done

    // Issue next tile's global loads early (land while we do math)
    const bool next = (k0 + 32) < K;
    uint4 a0, a1, bvv;
    if (next) {
      const uint4* ag =
          (const uint4*)(A + (size_t)(rowBase + ar) * K + (k0 + 32) + ac);
      a0 = ag[0];
      a1 = ag[1];
      bvv = *(const uint4*)(Bm + (size_t)(k0 + 32 + bk) * N + colBase + bn);
      if (k0 + 64 < K) {
        __builtin_prefetch(A + (size_t)(rowBase + ar) * K + k0 + 64 + ac, 0, 1);
        __builtin_prefetch(Bm + (size_t)(k0 + 64 + bk) * N + colBase + bn, 0, 1);
      }
    }

    // A frag: lane M=ln holds k = 16h..16h+15
    FragBF aF[2], bF[2];
#pragma unroll
    for (int mi = 0; mi < 2; mi++) {
      const unsigned short* ap =
          &sA[buf][(wm * 32 + mi * 16 + ln) * 32 + 16 * half];
      aF[mi].u[0] = *(const uint4*)ap;
      aF[mi].u[1] = *(const uint4*)(ap + 8);
    }
    // B frag: n = ln, lanes<16 hold k=0..15, lanes>=16 hold k=16..31
#pragma unroll
    for (int ni = 0; ni < 2; ni++) {
      const unsigned short* bp =
          &sBt[buf][(wn * 32 + ni * 16 + ln) * 32 + 16 * half];
      bF[ni].u[0] = *(const uint4*)bp;
      bF[ni].u[1] = *(const uint4*)(bp + 8);
    }
#pragma unroll
    for (int mi = 0; mi < 2; mi++)
#pragma unroll
      for (int ni = 0; ni < 2; ni++)
        acc[mi][ni] = wmma_bf16(aF[mi].v, bF[ni].v, acc[mi][ni]);

    // Store next tile into the other buffer
    if (next) {
      const int ob = buf ^ 1;
      *(uint4*)&sA[ob][ar * 32 + ac]     = a0;
      *(uint4*)&sA[ob][ar * 32 + ac + 8] = a1;
      const unsigned short* bs = (const unsigned short*)&bvv;
#pragma unroll
      for (int j = 0; j < 8; j++) sBt[ob][(bn + j) * 32 + bk] = bs[j];
    }
    buf ^= 1;
  }

  // C layout: element g -> row g + 8*half, col = ln
#pragma unroll
  for (int mi = 0; mi < 2; mi++)
#pragma unroll
    for (int ni = 0; ni < 2; ni++)
#pragma unroll
      for (int g = 0; g < 8; g++) {
        const int r = rowBase + wm * 32 + mi * 16 + g + 8 * half;
        const int c = colBase + wn * 32 + ni * 16 + ln;
        C[(size_t)r * N + c] = acc[mi][ni][g];
      }
}

// ---------------------------------------------------------------------------
// Fused L2-norm + RoPE, fp32 [B*T, H*D] -> bf16 [B, H, T, D]
// One wave per (b,t,h) row; lane owns d = {lane, lane+32, lane+64, lane+96}
// ---------------------------------------------------------------------------
__global__ __launch_bounds__(256) void norm_rope_kernel(
    const float* __restrict__ src, unsigned short* __restrict__ dst, int H) {
  const int lane = threadIdx.x & 31, wave = threadIdx.x >> 5;
  const int wg = blockIdx.x * 8 + wave;
  const int r = wg / H, h = wg % H;     // r = b*T + t
  const int t = r % T_;

  const float* sp = src + (size_t)r * (H * D_) + h * D_;
  float v0 = sp[lane], v1 = sp[lane + 32], v2 = sp[lane + 64], v3 = sp[lane + 96];
  float ss = v0 * v0 + v1 * v1 + v2 * v2 + v3 * v3;
  ss += __shfl_xor(ss, 1);  ss += __shfl_xor(ss, 2);
  ss += __shfl_xor(ss, 4);  ss += __shfl_xor(ss, 8);
  ss += __shfl_xor(ss, 16);
  const float inv = 1.f / fmaxf(sqrtf(ss), 1e-12f);
  v0 *= inv; v1 *= inv; v2 *= inv; v3 *= inv;

  // rotate-half RoPE: pairs (lane, lane+64) and (lane+32, lane+96)
  const float f0 = (float)t * __powf(10000.f, -(float)lane / 64.f);
  const float f1 = (float)t * __powf(10000.f, -(float)(lane + 32) / 64.f);
  const float c0 = __cosf(f0), s0 = __sinf(f0);
  const float c1 = __cosf(f1), s1 = __sinf(f1);
  const float o0 = v0 * c0 - v2 * s0;
  const float o2 = v2 * c0 + v0 * s0;
  const float o1 = v1 * c1 - v3 * s1;
  const float o3 = v3 * c1 + v1 * s1;

  unsigned short* dp = dst + (((size_t)(r / T_) * H + h) * T_ + t) * D_;
  dp[lane]      = f2bf(o0);
  dp[lane + 32] = f2bf(o1);
  dp[lane + 64] = f2bf(o2);
  dp[lane + 96] = f2bf(o3);
}

// V: fp32 [B*T, NKV*D] -> bf16 [B, NKV, T, D]
__global__ void v_pack_kernel(const float* __restrict__ vf,
                              unsigned short* __restrict__ vbuf) {
  const int idx = blockIdx.x * blockDim.x + threadIdx.x;
  const int cols = NKV_ * D_;
  const int r = idx / cols, c = idx % cols;
  const int hh = c / D_, d = c % D_;
  const int bb = r / T_, tt = r % T_;
  vbuf[(((size_t)bb * NKV_ + hh) * T_ + tt) * D_ + d] = f2bf(vf[idx]);
}

// ---------------------------------------------------------------------------
// Flash attention with sliding window, WMMA bf16.
// Grid: (T/128, NH, B). 8 waves; wave w owns query rows r0..r0+15.
// K tile (32x128 bf16, 8KB) is fetched by the Tensor Data Mover (wave 0
// issues tensor_load_to_lds, waits on TENSORcnt); V is staged transposed
// by all threads. Key loop over 32-key tiles of the block's window union.
// ---------------------------------------------------------------------------
__global__ __launch_bounds__(256) void swa_attn_wmma(
    const unsigned short* __restrict__ qb, const unsigned short* __restrict__ kb,
    const unsigned short* __restrict__ vb, unsigned short* __restrict__ ab) {
  __shared__ unsigned short sK[32 * 128];    // [key][d]   (TDM destination)
  __shared__ unsigned short sVt[128 * 32];   // [d][key]
  __shared__ unsigned short sP[8][16 * 32];  // per-wave P relayout buffer

  const int tid  = threadIdx.x;
  const int lane = tid & 31, wave = tid >> 5;
  const int half = lane >> 4, ln = lane & 15;
  const int R0 = blockIdx.x * 128;
  const int h  = blockIdx.y;
  const int b  = blockIdx.z;
  const int kvh = h >> 2;                    // NH/NKV = 4
  const int r0 = R0 + wave * 16;

  // Preload Q fragments (16 rows x 128 d) straight from global in A-layout
  FragBF aQ[4];
  {
    const unsigned short* qp =
        qb + ((size_t)(b * NH_ + h) * T_ + r0 + ln) * D_;
#pragma unroll
    for (int c = 0; c < 4; c++) {
      aQ[c].u[0] = *(const uint4*)(qp + 32 * c + 16 * half);
      aQ[c].u[1] = *(const uint4*)(qp + 32 * c + 8 + 16 * half);
    }
  }

  v8f zf = {};
  v8f o[8];
#pragma unroll
  for (int nt = 0; nt < 8; nt++) o[nt] = zf;
  float mrow[8], lrow[8];
#pragma unroll
  for (int g = 0; g < 8; g++) { mrow[g] = -1e30f; lrow[g] = 0.f; }

  const int js = R0 - WIN_ + 1;
  const int jstart = (js > 0 ? js : 0) & ~31;
  const unsigned short* kbase = kb + ((size_t)(b * NKV_ + kvh) * T_) * D_;
  const unsigned short* vbase = vb + ((size_t)(b * NKV_ + kvh) * T_) * D_;

  // Static part of the Tensor DMA descriptor (D# group 1):
  //   2D tile: tile_dim0 = 128 elems (row of d), tile_dim1 = 32 keys,
  //   data_size = 2B, tensor_dim0 = 128, tensor_dim1 = T, stride0 = 128.
  const u32x8 tdmG1 = {
      (1u << 16),                    // workgroup_mask=0, data_size=1 (2B)
      (128u << 16),                  // tensor_dim0[15:0] in [31:16]
      ((unsigned)T_ << 16),          // tensor_dim0 hi=0 | tensor_dim1 lo16
      (128u << 16),                  // tensor_dim1 hi=0 | tile_dim0=128
      32u,                           // tile_dim1=32, tile_dim2=0
      128u,                          // tensor_dim0_stride lo32 = 128
      0u, 0u};                       // stride hi / dim1_stride (unused, 2D)
  const unsigned sK_lds = lds_offset_of(sK);

  const int lr = tid >> 3;           // 0..31  (V row stager)
  const int lc = (tid & 7) * 16;     // 0..112 (V d-column stager)

  for (int j0 = jstart; j0 < R0 + 128; j0 += 32) {
    // --- K tile via Tensor Data Mover (wave 0 only) ---
    if (wave == 0) {
      const unsigned long long ga =
          (unsigned long long)(const void*)(kbase + (size_t)j0 * D_);
      u32x4 g0;
      g0.x = 1u;                                             // count=1
      g0.y = sK_lds;                                         // lds_addr
      g0.z = (unsigned)ga;                                   // global_addr lo
      g0.w = ((unsigned)(ga >> 32) & 0x01FFFFFFu) | (2u << 30);  // hi | type=2
      asm volatile("tensor_load_to_lds %0, %1"
                   :: "s"(g0), "s"(tdmG1) : "memory");
    }

    // --- V^T tile staged by all threads ---
    {
      const uint4* vg = (const uint4*)(vbase + (size_t)(j0 + lr) * D_ + lc);
      uint4 v0v = vg[0], v1v = vg[1];
      const unsigned short* vs0 = (const unsigned short*)&v0v;
      const unsigned short* vs1 = (const unsigned short*)&v1v;
#pragma unroll
      for (int j = 0; j < 8; j++) sVt[(lc + j) * 32 + lr]     = vs0[j];
#pragma unroll
      for (int j = 0; j < 8; j++) sVt[(lc + 8 + j) * 32 + lr] = vs1[j];
    }
    if (wave == 0) __builtin_amdgcn_s_wait_tensorcnt(0);
    __syncthreads();

    // S = Q * K^T  -> two 16x16 f32 tiles (keys j0..j0+15, j0+16..j0+31)
    v8f s0 = zf, s1 = zf;
#pragma unroll
    for (int c = 0; c < 4; c++) {
      FragBF bK0, bK1;
      const unsigned short* kp0 = &sK[ln * 128 + 32 * c + 16 * half];
      const unsigned short* kp1 = &sK[(16 + ln) * 128 + 32 * c + 16 * half];
      bK0.u[0] = *(const uint4*)kp0; bK0.u[1] = *(const uint4*)(kp0 + 8);
      bK1.u[0] = *(const uint4*)kp1; bK1.u[1] = *(const uint4*)(kp1 + 8);
      s0 = wmma_bf16(aQ[c].v, bK0.v, s0);
      s1 = wmma_bf16(aQ[c].v, bK1.v, s1);
    }

    // Online softmax (C layout: element g -> query row g+8*half, col = ln)
    unsigned short* myP = sP[wave];
#pragma unroll
    for (int g = 0; g < 8; g++) {
      const int qrow = r0 + g + 8 * half;
      const int c0 = j0 + ln, c1 = j0 + 16 + ln;
      const bool a0 = (c0 <= qrow) && (c0 + WIN_ > qrow);
      const bool a1 = (c1 <= qrow) && (c1 + WIN_ > qrow);
      const float v0 = s0[g] * SCALE_, v1 = s1[g] * SCALE_;
      float x = fmaxf(a0 ? v0 : -1e30f, a1 ? v1 : -1e30f);
      x = fmaxf(x, __shfl_xor(x, 1));
      x = fmaxf(x, __shfl_xor(x, 2));
      x = fmaxf(x, __shfl_xor(x, 4));
      x = fmaxf(x, __shfl_xor(x, 8));
      const float mn    = fmaxf(mrow[g], x);
      const float alpha = __expf(mrow[g] - mn);
      const float p0 = a0 ? __expf(v0 - mn) : 0.f;
      const float p1 = a1 ? __expf(v1 - mn) : 0.f;
      float rs = p0 + p1;
      rs += __shfl_xor(rs, 1);
      rs += __shfl_xor(rs, 2);
      rs += __shfl_xor(rs, 4);
      rs += __shfl_xor(rs, 8);
      lrow[g] = alpha * lrow[g] + rs;
      mrow[g] = mn;
#pragma unroll
      for (int nt = 0; nt < 8; nt++) o[nt][g] *= alpha;
      myP[(g + 8 * half) * 32 + ln]      = f2bf(p0);
      myP[(g + 8 * half) * 32 + 16 + ln] = f2bf(p1);
    }

    // Re-read P in A layout (16x32) — same-wave LDS ops are ordered
    FragBF aP;
    {
      const unsigned short* pp = &myP[ln * 32 + 16 * half];
      aP.u[0] = *(const uint4*)pp;
      aP.u[1] = *(const uint4*)(pp + 8);
    }

    // O += P * V  (8 x 16-wide d tiles)
#pragma unroll
    for (int nt = 0; nt < 8; nt++) {
      FragBF bV;
      const unsigned short* vp = &sVt[(nt * 16 + ln) * 32 + 16 * half];
      bV.u[0] = *(const uint4*)vp;
      bV.u[1] = *(const uint4*)(vp + 8);
      o[nt] = wmma_bf16(aP.v, bV.v, o[nt]);
    }
    __syncthreads();
  }

  // Normalize and store bf16 into [B*T, NH*D] for the output GEMM
#pragma unroll
  for (int g = 0; g < 8; g++) {
    const float inv = 1.f / fmaxf(lrow[g], 1e-20f);
    const size_t rowoff =
        ((size_t)b * T_ + r0 + g + 8 * half) * (NH_ * D_) + (size_t)h * D_;
#pragma unroll
    for (int nt = 0; nt < 8; nt++)
      ab[rowoff + nt * 16 + ln] = f2bf(o[nt][g] * inv);
  }
}

// ---------------------------------------------------------------------------
extern "C" void kernel_launch(void* const* d_in, const int* in_sizes, int n_in,
                              void* d_out, int out_size, void* d_ws,
                              size_t ws_size, hipStream_t stream) {
  (void)in_sizes; (void)n_in; (void)out_size; (void)ws_size;
  const float* x  = (const float*)d_in[0];
  const float* wq = (const float*)d_in[1];
  const float* wk = (const float*)d_in[2];
  const float* wv = (const float*)d_in[3];
  const float* wo = (const float*)d_in[4];
  float* out = (float*)d_out;

  char* base = (char*)d_ws;
  size_t off = 0;
  auto alloc = [&](size_t bytes) -> void* {
    void* r = base + off;
    off += bytes;
    off = (off + 255) & ~(size_t)255;
    return r;
  };

  unsigned short* xb  = (unsigned short*)alloc((size_t)MROWS * HID_ * 2);
  unsigned short* wqb = (unsigned short*)alloc((size_t)HID_ * NH_ * D_ * 2);
  unsigned short* wkb = (unsigned short*)alloc((size_t)HID_ * NKV_ * D_ * 2);
  unsigned short* wvb = (unsigned short*)alloc((size_t)HID_ * NKV_ * D_ * 2);
  unsigned short* wob = (unsigned short*)alloc((size_t)NH_ * D_ * HID_ * 2);
  float* qf = (float*)alloc((size_t)MROWS * NH_ * D_ * 4);
  float* kf = (float*)alloc((size_t)MROWS * NKV_ * D_ * 4);
  float* vf = (float*)alloc((size_t)MROWS * NKV_ * D_ * 4);
  unsigned short* qbb = (unsigned short*)alloc((size_t)MROWS * NH_ * D_ * 2);
  unsigned short* kbb = (unsigned short*)alloc((size_t)MROWS * NKV_ * D_ * 2);
  unsigned short* vbb = (unsigned short*)alloc((size_t)MROWS * NKV_ * D_ * 2);
  unsigned short* ab  = (unsigned short*)alloc((size_t)MROWS * NH_ * D_ * 2);

  // 1) fp32 -> bf16 casts
  cast_bf16_kernel<<<2048, 256, 0, stream>>>(x,  xb,  MROWS * HID_);
  cast_bf16_kernel<<<2048, 256, 0, stream>>>(wq, wqb, HID_ * NH_ * D_);
  cast_bf16_kernel<<<1024, 256, 0, stream>>>(wk, wkb, HID_ * NKV_ * D_);
  cast_bf16_kernel<<<1024, 256, 0, stream>>>(wv, wvb, HID_ * NKV_ * D_);
  cast_bf16_kernel<<<2048, 256, 0, stream>>>(wo, wob, NH_ * D_ * HID_);

  // 2) QKV projections (WMMA)
  gemm_bf16_wmma<<<dim3(2048 / 64, MROWS / 128), 256, 0, stream>>>(
      xb, wqb, qf, MROWS, 2048, HID_);
  gemm_bf16_wmma<<<dim3(512 / 64, MROWS / 128), 256, 0, stream>>>(
      xb, wkb, kf, MROWS, 512, HID_);
  gemm_bf16_wmma<<<dim3(512 / 64, MROWS / 128), 256, 0, stream>>>(
      xb, wvb, vf, MROWS, 512, HID_);

  // 3) l2norm + RoPE -> bf16 [B,H,T,D]; V pack
  norm_rope_kernel<<<(MROWS * NH_) / 8, 256, 0, stream>>>(qf, qbb, NH_);
  norm_rope_kernel<<<(MROWS * NKV_) / 8, 256, 0, stream>>>(kf, kbb, NKV_);
  v_pack_kernel<<<(MROWS * NKV_ * D_) / 256, 256, 0, stream>>>(vf, vbb);

  // 4) sliding-window flash attention (WMMA + TDM)
  swa_attn_wmma<<<dim3(T_ / 128, NH_, B_), 256, 0, stream>>>(qbb, kbb, vbb, ab);

  // 5) output projection (WMMA) -> f32
  gemm_bf16_wmma<<<dim3(HID_ / 64, MROWS / 128), 256, 0, stream>>>(
      ab, wob, out, MROWS, HID_, NH_ * D_);
}
